// entropy_18339510354594
// MI455X (gfx1250) — matrix operand order repaired and checked by reading, not compiled, not run on hardware
//
#include <hip/hip_runtime.h>
#include <hip/hip_bf16.h>
#include <math.h>

// ---------------------------------------------------------------------------
// Fused cosine-similarity GEMM + row/col max + entropy, MI455X (gfx1250).
//   C[n,m] = dot(ex[n], ey[m]) / max(|ex[n]|*|ey[m]|, 1e-8); entropy of
//   Normal log-prob of (1 + rowmax) and (1 + colmax).
// Compute-bound (AI ~2000 FLOP/byte). Strategy:
//   phase 1: split fp32 -> bf16 hi/lo once (ey also transposed to [d][m])
//   phase 2: bf16x3 WMMA GEMM (hi*hi + hi*lo + lo*hi), fp32 accum,
//            double-buffered LDS fed by global_load_async_to_lds_b128,
//            fused row/col max epilogue (C never materialized)
//   phase 3: entropy reduction over the 2x8192 maxima
// Fallback (small ws): single-kernel fused conversion GEMM from round 0.
// ---------------------------------------------------------------------------

typedef __attribute__((ext_vector_type(16))) __bf16 v16bf;
typedef __attribute__((ext_vector_type(8)))  __bf16 v8bf;
typedef __attribute__((ext_vector_type(4)))  __bf16 v4bf;
typedef __attribute__((ext_vector_type(8)))  float  v8f;

#define NROWS 8192
#define DDIM  768
#define EPS   1e-8f
// -(log(0.3) + 0.5*log(2*pi))
#define NEG_LOG_NORM_CONST (0.28503427112126335f)

// fast path tiles
#define FBM 128
#define FBN 128
#define FBK 32
#define KTILES (DDIM / FBK)   // 24

// fallback tiles
#define BM 128
#define BN 64
#define BK 32

#define WAIT_ASYNC(n) asm volatile("s_wait_asynccnt " #n ::: "memory")

// order-preserving float <-> uint for atomic max
__device__ __forceinline__ unsigned encf(float f) {
  unsigned b = __float_as_uint(f);
  return (b & 0x80000000u) ? ~b : (b | 0x80000000u);
}
__device__ __forceinline__ float decf(unsigned u) {
  unsigned b = (u & 0x80000000u) ? (u & 0x7FFFFFFFu) : ~u;
  return __uint_as_float(b);
}

__device__ __forceinline__ void split_bf16(float f, __bf16& hi, __bf16& lo) {
  hi = (__bf16)f;
  lo = (__bf16)(f - (float)hi);
}

__device__ __forceinline__ unsigned lds_addr(const void* p) {
  // flat shared-aperture address: low 32 bits are the LDS byte offset
  return (unsigned)(unsigned long long)p;
}

__device__ __forceinline__ void async_copy_b128(unsigned lds, unsigned goff,
                                                const __bf16* base) {
  asm volatile("global_load_async_to_lds_b128 %0, %1, %2"
               :: "v"(lds), "v"(goff), "s"(base) : "memory");
}

// --------------------------- row norms (both matrices) ---------------------
__global__ void norms_kernel(const float* __restrict__ ex,
                             const float* __restrict__ ey,
                             float* __restrict__ nx, float* __restrict__ ny) {
  int wave = threadIdx.x >> 5;
  int lane = threadIdx.x & 31;
  int row  = blockIdx.x * 8 + wave;
  const float* src = ex;
  float*       dst = nx;
  int r = row;
  if (row >= NROWS) { src = ey; dst = ny; r = row - NROWS; }
  float s = 0.0f;
  const float* p = src + (size_t)r * DDIM;
  for (int i = lane; i < DDIM; i += 32) {
    float v = p[i];
    s += v * v;
  }
#pragma unroll
  for (int off = 16; off > 0; off >>= 1) s += __shfl_xor(s, off, 32);
  if (lane == 0) dst[r] = sqrtf(s);
}

// --------------------------- phase 1: bf16 hi/lo split ---------------------
__global__ void convert_split_kernel(const float* __restrict__ src,
                                     __bf16* __restrict__ hi,
                                     __bf16* __restrict__ lo) {
  size_t i4 = ((size_t)blockIdx.x * 256 + threadIdx.x) * 4;
  float4 v = *(const float4*)(src + i4);
  __bf16 h0, h1, h2, h3, l0, l1, l2, l3;
  split_bf16(v.x, h0, l0); split_bf16(v.y, h1, l1);
  split_bf16(v.z, h2, l2); split_bf16(v.w, h3, l3);
  *(v4bf*)(hi + i4) = (v4bf){h0, h1, h2, h3};
  *(v4bf*)(lo + i4) = (v4bf){l0, l1, l2, l3};
}

// ey [m][d] -> eyT hi/lo [d][m] (64x64 tiles via LDS)
__global__ void convert_transpose_kernel(const float* __restrict__ ey,
                                         __bf16* __restrict__ hiT,
                                         __bf16* __restrict__ loT) {
  __shared__ float tile[64][65];
  const int t  = threadIdx.x;
  const int d0 = blockIdx.x * 64;
  const int m0 = blockIdx.y * 64;
#pragma unroll
  for (int i = 0; i < 4; ++i) {
    int c = t + 256 * i;
    int r = c >> 4;             // m within tile
    int cc = (c & 15) * 4;      // d within tile
    float4 v = *(const float4*)(ey + (size_t)(m0 + r) * DDIM + d0 + cc);
    tile[r][cc + 0] = v.x; tile[r][cc + 1] = v.y;
    tile[r][cc + 2] = v.z; tile[r][cc + 3] = v.w;
  }
  __syncthreads();
#pragma unroll
  for (int i = 0; i < 4; ++i) {
    int c  = t + 256 * i;
    int rd = c >> 4;            // d within tile
    int mc = (c & 15) * 4;      // m within tile
    __bf16 h[4], l[4];
#pragma unroll
    for (int j = 0; j < 4; ++j) split_bf16(tile[mc + j][rd], h[j], l[j]);
    size_t o = (size_t)(d0 + rd) * NROWS + m0 + mc;
    *(v4bf*)(hiT + o) = (v4bf){h[0], h[1], h[2], h[3]};
    *(v4bf*)(loT + o) = (v4bf){l[0], l[1], l[2], l[3]};
  }
}

// --------------------------- phase 2: async WMMA GEMM ----------------------
__global__ void __launch_bounds__(256)
cosmax_gemm_async_kernel(const __bf16* __restrict__ gxhi, const __bf16* __restrict__ gxlo,
                         const __bf16* __restrict__ gyhiT, const __bf16* __restrict__ gyloT,
                         const float* __restrict__ nx, const float* __restrict__ ny,
                         unsigned* __restrict__ rowBits, unsigned* __restrict__ colBits) {
  // double-buffered bf16 tiles: A [row][k], B already [k][n]
  __shared__ __align__(16) __bf16 sA[2][2][FBM * FBK];   // [buf][hi/lo]
  __shared__ __align__(16) __bf16 sB[2][2][FBK * FBN];
  __shared__ float    sNx[FBM];
  __shared__ float    sNy[FBN];
  __shared__ unsigned sRowMax[FBM];
  __shared__ unsigned sColMax[FBN];

  const int t      = threadIdx.x;
  const int lane   = t & 31;
  const int waveId = t >> 5;               // 8 waves: 4 (M) x 2 (N)
  const int wm     = (waveId >> 1) * 32;   // wave tile 32 x 64
  const int wn     = (waveId & 1) * 64;
  const int mbase  = blockIdx.x * FBM;
  const int nbase  = blockIdx.y * FBN;

  if (t < FBM) { sNx[t] = nx[mbase + t]; sRowMax[t] = 0u; }
  if (t < FBN) { sNy[t] = ny[nbase + t]; sColMax[t] = 0u; }

  // stage one K-tile into buffer `buf` (8 async b128 copies per thread)
  auto stage = [&](int kt, int buf) {
    int k0 = kt * FBK;
#pragma unroll
    for (int i = 0; i < 2; ++i) {          // A: 512 b128 chunks per array
      int c   = t + 256 * i;
      int row = c >> 2;
      int kc  = (c & 3) * 8;
      unsigned goff = (unsigned)(((mbase + row) * DDIM + k0 + kc) * 2);
      async_copy_b128(lds_addr(&sA[buf][0][row * FBK + kc]), goff, gxhi);
      async_copy_b128(lds_addr(&sA[buf][1][row * FBK + kc]), goff, gxlo);
    }
#pragma unroll
    for (int i = 0; i < 2; ++i) {          // B: 512 b128 chunks per array
      int c  = t + 256 * i;
      int kr = c >> 4;
      int nc = (c & 15) * 8;
      unsigned goff = (unsigned)(((k0 + kr) * NROWS + nbase + nc) * 2);
      async_copy_b128(lds_addr(&sB[buf][0][kr * FBN + nc]), goff, gyhiT);
      async_copy_b128(lds_addr(&sB[buf][1][kr * FBN + nc]), goff, gyloT);
    }
  };

  v8f acc[2][4] = {};                      // 32x64 wave tile, fp32 accum

  stage(0, 0);
  for (int kt = 0; kt < KTILES; ++kt) {
    const int buf = kt & 1;
    __syncthreads();                       // prev compute done reading buf^1
    if (kt + 1 < KTILES) {
      stage(kt + 1, buf ^ 1);
      WAIT_ASYNC(8);                       // in-order: oldest 8 (tile kt) done
    } else {
      WAIT_ASYNC(0);
    }
    __syncthreads();                       // tile kt visible to all waves

    // A 16-bit 16x32 frag: lane L -> row (L&15); V0-3=K[c..c+7],
    // V4-7=K[c+16..c+23], c=(L>>4)*8
    union V16 { v16bf v; v8bf h[2]; };
    V16 ahi[2], alo[2];
    const int c8 = (lane >> 4) << 3;
#pragma unroll
    for (int tm = 0; tm < 2; ++tm) {
      int row = wm + tm * 16 + (lane & 15);
      ahi[tm].h[0] = *(const v8bf*)&sA[buf][0][row * FBK + c8];
      ahi[tm].h[1] = *(const v8bf*)&sA[buf][0][row * FBK + c8 + 16];
      alo[tm].h[0] = *(const v8bf*)&sA[buf][1][row * FBK + c8];
      alo[tm].h[1] = *(const v8bf*)&sA[buf][1][row * FBK + c8 + 16];
    }
    // B 16-bit 32x16 frag: lane L -> col (L&15); same K order as A
    v16bf bhi[4], blo[4];
#pragma unroll
    for (int tn = 0; tn < 4; ++tn) {
#pragma unroll
      for (int s = 0; s < 16; ++s) {
        int kk = (s < 8) ? (c8 + s) : (16 + c8 + s - 8);
        bhi[tn][s] = sB[buf][0][kk * FBN + wn + tn * 16 + (lane & 15)];
        blo[tn][s] = sB[buf][1][kk * FBN + wn + tn * 16 + (lane & 15)];
      }
    }
    // 24 WMMAs: hi*hi + hi*lo + lo*hi
#pragma unroll
    for (int tm = 0; tm < 2; ++tm) {
#pragma unroll
      for (int tn = 0; tn < 4; ++tn) {
        acc[tm][tn] = __builtin_amdgcn_wmma_f32_16x16x32_bf16(
            false, ahi[tm].v, false, bhi[tn], (short)0, acc[tm][tn], false, false);
        acc[tm][tn] = __builtin_amdgcn_wmma_f32_16x16x32_bf16(
            false, ahi[tm].v, false, blo[tn], (short)0, acc[tm][tn], false, false);
        acc[tm][tn] = __builtin_amdgcn_wmma_f32_16x16x32_bf16(
            false, alo[tm].v, false, bhi[tn], (short)0, acc[tm][tn], false, false);
      }
    }
  }
  __syncthreads();

  // epilogue: C/D layout: lane L, VGPR j -> M = j + 8*(L>>4), N = L&15
#pragma unroll
  for (int tm = 0; tm < 2; ++tm) {
#pragma unroll
    for (int tn = 0; tn < 4; ++tn) {
      int cn    = wn + tn * 16 + (lane & 15);
      int rbase = wm + tm * 16 + ((lane >> 4) << 3);
      float nyv = sNy[cn];
      float cmax = -3.0e38f;
#pragma unroll
      for (int j = 0; j < 8; ++j) {
        int r = rbase + j;
        float den = fmaxf(sNx[r] * nyv, EPS);
        float cv  = acc[tm][tn][j] / den;
        atomicMax(&sRowMax[r], encf(cv));
        cmax = fmaxf(cmax, cv);
      }
      atomicMax(&sColMax[cn], encf(cmax));
    }
  }
  __syncthreads();

  if (t < FBM) {
    atomicMax(&rowBits[mbase + t], sRowMax[t]);
  } else {
    atomicMax(&colBits[nbase + t - FBM], sColMax[t - FBM]);
  }
}

// --------------------------- fallback fused GEMM (small ws) ----------------
__global__ void __launch_bounds__(256)
cosmax_gemm_kernel(const float* __restrict__ ex, const float* __restrict__ ey,
                   const float* __restrict__ nx, const float* __restrict__ ny,
                   unsigned* __restrict__ rowBits, unsigned* __restrict__ colBits) {
  __shared__ __align__(16) __bf16 sAhi[BM * BK];
  __shared__ __align__(16) __bf16 sAlo[BM * BK];
  __shared__ __align__(16) __bf16 sBhi[BK * BN];   // [k][n]
  __shared__ __align__(16) __bf16 sBlo[BK * BN];
  __shared__ float    sNx[BM];
  __shared__ float    sNy[BN];
  __shared__ unsigned sRowMax[BM];
  __shared__ unsigned sColMax[BN];

  const int t      = threadIdx.x;
  const int lane   = t & 31;
  const int waveId = t >> 5;
  const int wm     = (waveId >> 1) * 32;
  const int wn     = (waveId & 1) * 32;
  const int mbase  = blockIdx.x * BM;
  const int nbase  = blockIdx.y * BN;

  if (t < BM) { sNx[t] = nx[mbase + t]; sRowMax[t] = 0u; }
  if (t < BN) { sNy[t] = ny[nbase + t]; sColMax[t] = 0u; }

  v8f acc[2][2] = {};

  for (int k0 = 0; k0 < DDIM; k0 += BK) {
    __syncthreads();
#pragma unroll
    for (int i = 0; i < 4; ++i) {
      int idx = t + 256 * i;
      int row = idx >> 3;
      int kc  = (idx & 7) << 2;
      const float4 v = *(const float4*)(ex + (size_t)(mbase + row) * DDIM + k0 + kc);
      __bf16 h0, h1, h2, h3, l0, l1, l2, l3;
      split_bf16(v.x, h0, l0); split_bf16(v.y, h1, l1);
      split_bf16(v.z, h2, l2); split_bf16(v.w, h3, l3);
      *(v4bf*)&sAhi[row * BK + kc] = (v4bf){h0, h1, h2, h3};
      *(v4bf*)&sAlo[row * BK + kc] = (v4bf){l0, l1, l2, l3};
    }
#pragma unroll
    for (int i = 0; i < 2; ++i) {
      int idx = t + 256 * i;
      int row = idx >> 3;
      int kc  = (idx & 7) << 2;
      const float4 v = *(const float4*)(ey + (size_t)(nbase + row) * DDIM + k0 + kc);
      __bf16 h, l;
      split_bf16(v.x, h, l); sBhi[(kc + 0) * BN + row] = h; sBlo[(kc + 0) * BN + row] = l;
      split_bf16(v.y, h, l); sBhi[(kc + 1) * BN + row] = h; sBlo[(kc + 1) * BN + row] = l;
      split_bf16(v.z, h, l); sBhi[(kc + 2) * BN + row] = h; sBlo[(kc + 2) * BN + row] = l;
      split_bf16(v.w, h, l); sBhi[(kc + 3) * BN + row] = h; sBlo[(kc + 3) * BN + row] = l;
    }
    __syncthreads();

    union V16 { v16bf v; v8bf h[2]; };
    V16 ahi[2], alo[2];
    const int c = (lane >> 4) << 3;
#pragma unroll
    for (int tm = 0; tm < 2; ++tm) {
      int row = wm + tm * 16 + (lane & 15);
      ahi[tm].h[0] = *(const v8bf*)&sAhi[row * BK + c];
      ahi[tm].h[1] = *(const v8bf*)&sAhi[row * BK + c + 16];
      alo[tm].h[0] = *(const v8bf*)&sAlo[row * BK + c];
      alo[tm].h[1] = *(const v8bf*)&sAlo[row * BK + c + 16];
    }
    v16bf bhi[2], blo[2];
#pragma unroll
    for (int tn = 0; tn < 2; ++tn) {
#pragma unroll
      for (int s = 0; s < 16; ++s) {
        int kk = (s < 8) ? (c + s) : (16 + c + s - 8);
        bhi[tn][s] = sBhi[kk * BN + wn + tn * 16 + (lane & 15)];
        blo[tn][s] = sBlo[kk * BN + wn + tn * 16 + (lane & 15)];
      }
    }
#pragma unroll
    for (int tm = 0; tm < 2; ++tm) {
#pragma unroll
      for (int tn = 0; tn < 2; ++tn) {
        acc[tm][tn] = __builtin_amdgcn_wmma_f32_16x16x32_bf16(
            false, ahi[tm].v, false, bhi[tn], (short)0, acc[tm][tn], false, false);
        acc[tm][tn] = __builtin_amdgcn_wmma_f32_16x16x32_bf16(
            false, ahi[tm].v, false, blo[tn], (short)0, acc[tm][tn], false, false);
        acc[tm][tn] = __builtin_amdgcn_wmma_f32_16x16x32_bf16(
            false, alo[tm].v, false, bhi[tn], (short)0, acc[tm][tn], false, false);
      }
    }
  }
  __syncthreads();

#pragma unroll
  for (int tm = 0; tm < 2; ++tm) {
#pragma unroll
    for (int tn = 0; tn < 2; ++tn) {
      int cn    = wn + tn * 16 + (lane & 15);
      int rbase = wm + tm * 16 + ((lane >> 4) << 3);
      float nyv = sNy[cn];
      float cmax = -3.0e38f;
#pragma unroll
      for (int j = 0; j < 8; ++j) {
        int r = rbase + j;
        float den = fmaxf(sNx[r] * nyv, EPS);
        float cv  = acc[tm][tn][j] / den;
        atomicMax(&sRowMax[r], encf(cv));
        cmax = fmaxf(cmax, cv);
      }
      atomicMax(&sColMax[cn], encf(cmax));
    }
  }
  __syncthreads();

  if (t < BM) {
    atomicMax(&rowBits[mbase + t], sRowMax[t]);
  } else if (t < BM + BN) {
    atomicMax(&colBits[nbase + t - BM], sColMax[t - BM]);
  }
}

// --------------------------- phase 3: entropy finalize ---------------------
__global__ void finalize_kernel(const unsigned* __restrict__ rowBits,
                                const unsigned* __restrict__ colBits,
                                float* __restrict__ out) {
  const unsigned* bits = (blockIdx.x == 0) ? rowBits : colBits;
  __shared__ float red[256];
  float s = 0.0f;
  for (int i = threadIdx.x; i < NROWS; i += 256) {
    float m = decf(bits[i]);
    float z = m * (1.0f / 0.3f);
    float cent = -0.5f * z * z + NEG_LOG_NORM_CONST;
    s += -(expf(cent) * cent);
  }
  red[threadIdx.x] = s;
  __syncthreads();
#pragma unroll
  for (int off = 128; off > 0; off >>= 1) {
    if (threadIdx.x < off) red[threadIdx.x] += red[threadIdx.x + off];
    __syncthreads();
  }
  if (threadIdx.x == 0) out[blockIdx.x] = red[0];
}

// --------------------------- launch ----------------------------------------
extern "C" void kernel_launch(void* const* d_in, const int* in_sizes, int n_in,
                              void* d_out, int out_size, void* d_ws, size_t ws_size,
                              hipStream_t stream) {
  const float* ex = (const float*)d_in[0];
  const float* ey = (const float*)d_in[1];
  float* out = (float*)d_out;

  char* ws = (char*)d_ws;
  float*    nx      = (float*)(ws + 0);
  float*    ny      = (float*)(ws + (size_t)NROWS * 4);
  unsigned* rowBits = (unsigned*)(ws + (size_t)NROWS * 8);
  unsigned* colBits = (unsigned*)(ws + (size_t)NROWS * 12);
  const size_t HDR  = (size_t)NROWS * 16;                 // 128 KB
  const size_t MAT  = (size_t)NROWS * DDIM * sizeof(__bf16);  // 12.58 MB

  hipMemsetAsync(rowBits, 0, (size_t)NROWS * 8, stream);  // enc(0) = -max
  norms_kernel<<<NROWS * 2 / 8, 256, 0, stream>>>(ex, ey, nx, ny);

  if (ws_size >= HDR + 4 * MAT) {
    __bf16* gxhi  = (__bf16*)(ws + HDR);
    __bf16* gxlo  = (__bf16*)(ws + HDR + MAT);
    __bf16* gyhiT = (__bf16*)(ws + HDR + 2 * MAT);
    __bf16* gyloT = (__bf16*)(ws + HDR + 3 * MAT);

    convert_split_kernel<<<NROWS * DDIM / 1024, 256, 0, stream>>>(ex, gxhi, gxlo);
    dim3 tgrid(DDIM / 64, NROWS / 64);
    convert_transpose_kernel<<<tgrid, 256, 0, stream>>>(ey, gyhiT, gyloT);

    dim3 grid(NROWS / FBM, NROWS / FBN);   // 64 x 64
    cosmax_gemm_async_kernel<<<grid, 256, 0, stream>>>(
        gxhi, gxlo, gyhiT, gyloT, nx, ny, rowBits, colBits);
  } else {
    dim3 grid(NROWS / BM, NROWS / BN);     // 64 x 128
    cosmax_gemm_kernel<<<grid, 256, 0, stream>>>(ex, ey, nx, ny, rowBits, colBits);
  }

  finalize_kernel<<<2, 256, 0, stream>>>(rowBits, colBits, out);
}
